// GCNEncoder_11982958756635
// MI455X (gfx1250) — matrix operand order repaired and checked by reading, not compiled, not run on hardware
//
#include <hip/hip_runtime.h>

// ---------------------------------------------------------------------------
// CDNA5 / gfx1250 2-layer GCN
//   - bf16 WMMA GEMMs (v_wmma_f32_16x16x32_bf16), operands pre-packed to bf16
//   - W pre-packed into per-lane fragment order -> B loads are 2x b128/step
//   - L2-resident float-atomic scatter aggregation
//   - bias fused into self-loop init (aggregation is additive)
// ---------------------------------------------------------------------------

typedef __bf16 v16bf __attribute__((ext_vector_type(16)));
typedef __bf16 v8bf  __attribute__((ext_vector_type(8)));
typedef float  v8f   __attribute__((ext_vector_type(8)));

// ---------------------- degree / normalization -----------------------------

__global__ __launch_bounds__(256) void deg_init(float* __restrict__ deg, int n) {
    int i = blockIdx.x * 256 + threadIdx.x;
    if (i < n) deg[i] = 1.0f;   // self-loop weight
}

__global__ __launch_bounds__(256) void deg_edges(const int* __restrict__ dst,
                                                 float* __restrict__ deg, int ne) {
    int e = blockIdx.x * 256 + threadIdx.x;
    if (e < ne) atomicAdd(&deg[dst[e]], 1.0f);
}

__global__ __launch_bounds__(256) void deg_rsqrt(float* __restrict__ deg, int n) {
    int i = blockIdx.x * 256 + threadIdx.x;
    if (i < n) deg[i] = rsqrtf(deg[i]);
}

// ---------------------- operand packing ------------------------------------

// fp32 -> bf16, 8 elements per thread (n8 = total/8)
__global__ __launch_bounds__(256)
void cvt_bf16(const float* __restrict__ in, __bf16* __restrict__ out, size_t n8) {
    size_t i = (size_t)blockIdx.x * 256 + threadIdx.x;
    if (i >= n8) return;
    const float4 f0 = ((const float4*)in)[2 * i];
    const float4 f1 = ((const float4*)in)[2 * i + 1];
    v8bf v;
    v[0] = (__bf16)f0.x; v[1] = (__bf16)f0.y; v[2] = (__bf16)f0.z; v[3] = (__bf16)f0.w;
    v[4] = (__bf16)f1.x; v[5] = (__bf16)f1.y; v[6] = (__bf16)f1.z; v[7] = (__bf16)f1.w;
    ((v8bf*)out)[i] = v;
}

// relu + fp32 -> bf16 (layer-1 activation feeding GEMM2)
__global__ __launch_bounds__(256)
void relu_bf16(const float* __restrict__ in, __bf16* __restrict__ out, size_t n8) {
    size_t i = (size_t)blockIdx.x * 256 + threadIdx.x;
    if (i >= n8) return;
    const float4 f0 = ((const float4*)in)[2 * i];
    const float4 f1 = ((const float4*)in)[2 * i + 1];
    v8bf v;
    v[0] = (__bf16)fmaxf(f0.x, 0.f); v[1] = (__bf16)fmaxf(f0.y, 0.f);
    v[2] = (__bf16)fmaxf(f0.z, 0.f); v[3] = (__bf16)fmaxf(f0.w, 0.f);
    v[4] = (__bf16)fmaxf(f1.x, 0.f); v[5] = (__bf16)fmaxf(f1.y, 0.f);
    v[6] = (__bf16)fmaxf(f1.z, 0.f); v[7] = (__bf16)fmaxf(f1.w, 0.f);
    ((v8bf*)out)[i] = v;
}

// Pack W[256,DOUT] (fp32, row-major) into bf16 per-lane B-fragment order:
//   Wp[ ((ct*8 + kb)*32 + lane)*16 + t ] = W[ (kb*32 + (t>>3)*16 + (lane>>4)*8 + (t&7))*DOUT
//                                             + ct*16 + (lane&15) ]
// so in the GEMM each lane's whole 16-element B fragment is one contiguous
// 32-byte load, coalesced across the wave.
template <int DOUT>
__global__ __launch_bounds__(256)
void pack_W(const float* __restrict__ W, __bf16* __restrict__ Wp) {
    const int idx = blockIdx.x * 256 + threadIdx.x;
    if (idx >= DOUT * 256) return;
    const int t    = idx & 15;
    const int lane = (idx >> 4) & 31;
    const int kb   = (idx >> 9) & 7;
    const int ct   = idx >> 12;
    const int K    = kb * 32 + (t >> 3) * 16 + (lane >> 4) * 8 + (t & 7);
    const int col  = ct * 16 + (lane & 15);
    Wp[idx] = (__bf16)W[(size_t)K * DOUT + col];
}

// ---------------------- WMMA GEMM : C[N,DOUT] = A[N,256] @ W[256,DOUT] ------
//
// blockDim = 256 (8 waves). Each wave owns one 16x16 f32 tile; K=256 in steps
// of 32 -> 8 v_wmma_f32_16x16x32_bf16 per wave. grid = (nrows/16, DOUT/128).
// A is bf16 row-major; B comes from the fragment-packed Wp.
//
// Fragment layouts per CDNA5 ISA 7.12.2 (wave32):
//   A (16x32 bf16): row = lane&15, a[t] -> K = k0 + (t>>3)*16 + (lane>>4)*8 + (t&7)
//   B (32x16 bf16): col = lane&15, b[t] -> K = k0 + (t>>3)*16 + (lane>>4)*8 + (t&7)
//   C (16x16 f32) : acc[r] -> row = (lane>>4)*8 + r, col = lane&15

template <int DOUT>
__global__ __launch_bounds__(256)
void gemm_wmma_bf16(const __bf16* __restrict__ A, const __bf16* __restrict__ Wp,
                    float* __restrict__ C, int nrows) {
    const int lane = threadIdx.x & 31;
    const int r16  = lane & 15;
    const int half = lane >> 4;
    const int wave = threadIdx.x >> 5;

    const int mbase   = blockIdx.x * 16;
    const int colTile = blockIdx.y * 8 + wave;
    const int cb      = colTile * 16;

    int arow = mbase + r16;
    if (arow >= nrows) arow = nrows - 1;                 // clamp (read-only safety)
    const __bf16* __restrict__ Arow = A + (size_t)arow * 256;
    const __bf16* __restrict__ Bp   = Wp + ((size_t)colTile * 8 * 32 + lane) * 16;

    v8f acc = {};
    #pragma unroll
    for (int k0 = 0; k0 < 256; k0 += 32) {
        const v8bf alo = *(const v8bf*)(Arow + k0 + half * 8);        // K = k0+h8+0..7
        const v8bf ahi = *(const v8bf*)(Arow + k0 + 16 + half * 8);   // K = k0+16+h8+0..7
        const v16bf a = __builtin_shufflevector(alo, ahi,
            0, 1, 2, 3, 4, 5, 6, 7, 8, 9, 10, 11, 12, 13, 14, 15);
        const v16bf b = *(const v16bf*)(Bp + (size_t)(k0 >> 5) * 32 * 16);
        acc = __builtin_amdgcn_wmma_f32_16x16x32_bf16(
            /*neg_a=*/false, a, /*neg_b=*/false, b,
            /*c_mod=*/(short)0, acc, /*reuse_a=*/false, /*reuse_b=*/false);
    }

    #pragma unroll
    for (int r = 0; r < 8; ++r) {
        const int row = mbase + half * 8 + r;
        if (row < nrows) C[(size_t)row * DOUT + cb + r16] = acc[r];
    }
}

// ---------------------- aggregation ----------------------------------------

// agg[i,d] = h[i,d]*dinv[i]^2 + b[d]   (self-loop term + fused bias; also
// initializes the atomic accumulation target)
template <int D>
__global__ __launch_bounds__(256)
void self_loop_bias(const float* __restrict__ h, const float* __restrict__ dinv,
                    const float* __restrict__ b, float* __restrict__ agg, int n) {
    size_t i = (size_t)blockIdx.x * 256 + threadIdx.x;
    if (i >= (size_t)n * D) return;
    const float di = dinv[i / D];
    agg[i] = h[i] * di * di + b[i & (D - 1)];
}

// One wave per edge: agg[dst,:] += h[src,:] * dinv[src]*dinv[dst]
// Coalesced 32-lane row traffic; atomic targets (<=150MB) stay L2-resident.
template <int D>
__global__ __launch_bounds__(256)
void edge_scatter(const float* __restrict__ h, const int* __restrict__ src,
                  const int* __restrict__ dst, const float* __restrict__ dinv,
                  float* __restrict__ agg, int ne) {
    const int lane = threadIdx.x & 31;
    const int e = blockIdx.x * 8 + (threadIdx.x >> 5);
    if (e >= ne) return;
    const int s = src[e];
    const int d = dst[e];
    const float norm = dinv[s] * dinv[d];
    const float* __restrict__ hs = h + (size_t)s * D;
    float* __restrict__ ad = agg + (size_t)d * D;
    #pragma unroll
    for (int i = lane; i < D; i += 32) {
        atomicAdd(&ad[i], hs[i] * norm);
    }
}

// ---------------------- launch ---------------------------------------------

extern "C" void kernel_launch(void* const* d_in, const int* in_sizes, int n_in,
                              void* d_out, int out_size, void* d_ws, size_t ws_size,
                              hipStream_t stream) {
    const float* x  = (const float*)d_in[0];
    const int*   ei = (const int*)d_in[1];
    const float* W1 = (const float*)d_in[2];
    const float* b1 = (const float*)d_in[3];
    const float* W2 = (const float*)d_in[4];
    const float* b2 = (const float*)d_in[5];
    float* out = (float*)d_out;

    const int N = in_sizes[0] / 256;   // 100000
    const int E = in_sizes[1] / 2;     // 1600000
    const int* src = ei;
    const int* dst = ei + E;

    // workspace layout (all 256B-aligned chunks):
    //   dinv[N] | bufA fp32[N*256] | bufB fp32[N*256] | abf bf16[N*256]
    //   | Wp1 bf16[65536] | Wp2 bf16[32768]              (~256.6 MB total)
    float*  dinv = (float*)d_ws;
    float*  bufA = dinv + (((size_t)N + 63) & ~(size_t)63);
    float*  bufB = bufA + (size_t)N * 256;
    __bf16* abf  = (__bf16*)(bufB + (size_t)N * 256);   // x(bf16), later h1(bf16)
    __bf16* Wp1  = abf + (size_t)N * 256;
    __bf16* Wp2  = Wp1 + 256 * 256;

    const dim3 blk(256);
    const unsigned nodeBlocks  = (unsigned)((N + 255) / 256);
    const unsigned edgeBlocksT = (unsigned)((E + 255) / 256);
    const unsigned edgeBlocksW = (unsigned)((E + 7) / 8);
    const unsigned elems256    = (unsigned)(((size_t)N * 256 + 255) / 256);
    const unsigned elems128    = (unsigned)(((size_t)N * 128 + 255) / 256);
    const unsigned vec256      = (unsigned)(((size_t)N * 256 / 8 + 255) / 256);
    const unsigned mTiles      = (unsigned)((N + 15) / 16);

    // normalization factors + operand packing (tiny)
    deg_init <<<nodeBlocks,  blk, 0, stream>>>(dinv, N);
    deg_edges<<<edgeBlocksT, blk, 0, stream>>>(dst, dinv, E);
    deg_rsqrt<<<nodeBlocks,  blk, 0, stream>>>(dinv, N);
    cvt_bf16 <<<vec256,      blk, 0, stream>>>(x, abf, (size_t)N * 256 / 8);
    pack_W<256><<<(256 * 256 + 255) / 256, blk, 0, stream>>>(W1, Wp1);
    pack_W<128><<<(128 * 256 + 255) / 256, blk, 0, stream>>>(W2, Wp2);

    // ---- layer 1: h1 = x@W1 ; agg1 = h1*dinv^2 + b1 + scatter ; relu->bf16
    gemm_wmma_bf16<256><<<dim3(mTiles, 2), blk, 0, stream>>>(abf, Wp1, bufA, N);
    self_loop_bias<256><<<elems256,    blk, 0, stream>>>(bufA, dinv, b1, bufB, N);
    edge_scatter<256>  <<<edgeBlocksW, blk, 0, stream>>>(bufA, src, dst, dinv, bufB, E);
    relu_bf16          <<<vec256,      blk, 0, stream>>>(bufB, abf, (size_t)N * 256 / 8);

    // ---- layer 2: h2 = h1@W2 ; out = h2*dinv^2 + b2 + scatter   (no epilogue)
    gemm_wmma_bf16<128><<<dim3(mTiles, 1), blk, 0, stream>>>(abf, Wp2, bufA, N);
    self_loop_bias<128><<<elems128,    blk, 0, stream>>>(bufA, dinv, b2, out, N);
    edge_scatter<128>  <<<edgeBlocksW, blk, 0, stream>>>(bufA, src, dst, dinv, out, E);
}
